// TAC_14774687498641
// MI455X (gfx1250) — compile-verified, matrix operand order checked by reading, not run on hardware
//
#include <hip/hip_runtime.h>

#define Bb   4
#define Gg   8
#define Nn   128
#define Tt   6000
#define Hh   384
#define BT   (Bb*Tt)        // 24000
#define EPSL 1e-8f

typedef float v8f   __attribute__((ext_vector_type(8)));
typedef __bf16 v16bf __attribute__((ext_vector_type(16)));
typedef unsigned short u16x4  __attribute__((ext_vector_type(4)));
typedef unsigned short u16x8  __attribute__((ext_vector_type(8)));
typedef unsigned short u16x16 __attribute__((ext_vector_type(16)));

__device__ __forceinline__ unsigned short f2bf(float f) {
    unsigned int u = __builtin_bit_cast(unsigned int, f);
    unsigned int r = u + 0x7FFFu + ((u >> 16) & 1u);   // round-to-nearest-even
    return (unsigned short)(r >> 16);
}

__device__ __forceinline__ float prelu(float x, float a) {
    return x >= 0.0f ? x : a * x;
}

// Async DMA one 8-byte chunk global -> LDS (CDNA5 GLOBAL_LOAD_ASYNC_TO_LDS_B64,
// tracked on ASYNCcnt; no VGPR round-trip).
__device__ __forceinline__ void async_copy_b64(unsigned int lds_off, const void* src) {
    asm volatile("global_load_async_to_lds_b64 %0, %1, off"
                 :: "v"(lds_off), "v"(src)
                 : "memory");
}
__device__ __forceinline__ void async_wait0() {
    asm volatile("s_wait_asynccnt 0" ::: "memory");
}

// Load a 16x32 bf16 WMMA fragment from LDS stored row-major-in-K.
// lane L<16: row r0+L, K = kk+{0..7, 16..23}; lane L>=16: K shifted by +8.
__device__ __forceinline__ v16bf frag_ld(const unsigned short* lds, int r0, int kk, int stride) {
    int lane = threadIdx.x & 31;
    int r  = r0 + (lane & 15);
    int kb = kk + ((lane & 16) >> 1);           // +8 for hi half-wave
    const unsigned short* p = lds + r * stride + kb;
    u16x8 lo = *(const u16x8*)p;
    u16x8 hi = *(const u16x8*)(p + 16);
    u16x16 c = __builtin_shufflevector(lo, hi, 0,1,2,3,4,5,6,7,8,9,10,11,12,13,14,15);
    return __builtin_bit_cast(v16bf, c);
}

__device__ __forceinline__ v8f wmma_bf16(v16bf a, v16bf b, v8f c) {
    return __builtin_amdgcn_wmma_f32_16x16x32_bf16(false, a, false, b, (short)0, c, false, false);
}

// ---------------- K0: zero stats ----------------
__global__ void tac_k0_zero(float* stats) {
    if (threadIdx.x < 64) stats[threadIdx.x] = 0.0f;
}

// ---------------- K1: GEMM1 (x -> gi) + group mean ----------------
// grid: (375 t-tiles of 16, 3 h-tiles of 128, 4 b). block 256 = 8 waves.
__global__ __launch_bounds__(256) void tac_k1_gemm_in(
    const float* __restrict__ x, const float* __restrict__ Wi,
    const float* __restrict__ bi, const float* __restrict__ aip,
    unsigned short* __restrict__ gi, unsigned short* __restrict__ gmean)
{
    __shared__ unsigned short Alds[128 * 128];
    __shared__ unsigned short Blds[128 * 128];
    const int t0  = blockIdx.x * 16;
    const int h0  = blockIdx.y * 128;
    const int b   = blockIdx.z;
    const int tid = threadIdx.x;
    const float ai = *aip;

    // A tile: rows r = t_local*8+g, cols k = n.  Read x[b,g,n,t0+tl] (f32) -> bf16.
    for (int i = 0; i < 16; ++i) {
        int f    = tid + i * 256;          // float4 index, 0..4095
        int quad = f & 3;                  // t sub-quad
        int n    = (f >> 2) & 127;
        int g    = f >> 9;
        const float4 v = *(const float4*)(x + ((size_t)(b * Gg + g) * Nn + n) * Tt + t0 + quad * 4);
        int base = (quad * 4 * Gg + g) * 128 + n;   // row stride per t_local = 8*128
        Alds[base + 0 * 1024] = f2bf(v.x);
        Alds[base + 1 * 1024] = f2bf(v.y);
        Alds[base + 2 * 1024] = f2bf(v.z);
        Alds[base + 3 * 1024] = f2bf(v.w);
    }
    // B tile: Blds[j][k] = Wi[h0+j][k]  (K-contiguous already)
    for (int i = 0; i < 16; ++i) {
        int f  = tid + i * 256;            // 0..4095
        int k4 = f & 31;
        int j  = f >> 5;
        const float4 v = *(const float4*)(Wi + (size_t)(h0 + j) * Nn + k4 * 4);
        u16x4 p; p.x = f2bf(v.x); p.y = f2bf(v.y); p.z = f2bf(v.z); p.w = f2bf(v.w);
        *(u16x4*)&Blds[j * 128 + k4 * 4] = p;
    }
    __syncthreads();

    const int wave = tid >> 5, lane = tid & 31;
    const int wm = wave >> 1, wn = wave & 1;
    const int m0 = wm * 32;
    const int j0 = wn * 64;
    v8f zero = {0,0,0,0,0,0,0,0};
    v8f acc[2][4];
    for (int im = 0; im < 2; ++im)
        for (int jn = 0; jn < 4; ++jn) acc[im][jn] = zero;

    for (int kk = 0; kk < 128; kk += 32) {
        v16bf a0 = frag_ld(Alds, m0,      kk, 128);
        v16bf a1 = frag_ld(Alds, m0 + 16, kk, 128);
        for (int jn = 0; jn < 4; ++jn) {
            v16bf bm_ = frag_ld(Blds, j0 + jn * 16, kk, 128);
            acc[0][jn] = wmma_bf16(a0, bm_, acc[0][jn]);
            acc[1][jn] = wmma_bf16(a1, bm_, acc[1][jn]);
        }
    }
    __syncthreads();   // done reading tiles; reuse Alds as D staging

    // epilogue: bias + PReLU -> stage bf16 D-tile in LDS; group mean direct
    const int colL = lane & 15;
    const int hiL  = (lane >> 4) & 1;
    for (int im = 0; im < 2; ++im) {
        int rbase   = m0 + im * 16 + hiL * 8;   // multiple of 8 -> g == v
        int t_local = rbase >> 3;
        for (int jn = 0; jn < 4; ++jn) {
            int col = j0 + jn * 16 + colL;
            int h = h0 + col;
            float bias = bi[h];
            float gsum = 0.0f;
            for (int v = 0; v < 8; ++v) {
                float val = prelu(acc[im][jn][v] + bias, ai);
                gsum += val;
                Alds[(rbase + v) * 128 + col] = f2bf(val);
            }
            gmean[((size_t)(b * Tt) + t0 + t_local) * Hh + h] = f2bf(gsum * 0.125f);
        }
    }
    __syncthreads();
    // coalesced copy-out: 2048 u16x8 (16B) stores
    for (int i = 0; i < 8; ++i) {
        int f = tid + i * 256;
        int c8 = f & 15, r = f >> 4;
        int tl = r >> 3, g = r & 7;
        size_t btg = ((size_t)(b * Tt + t0 + tl) * Gg) + g;
        *(u16x8*)(gi + btg * Hh + h0 + c8 * 8) = *(const u16x8*)&Alds[r * 128 + c8 * 8];
    }
}

// ---------------- K2: GEMM2 (gmean -> gm) ----------------
// grid: (188 row-tiles of 128, 3 n-tiles of 128). block 256.
__global__ __launch_bounds__(256) void tac_k2_gemm_mean(
    const unsigned short* __restrict__ gmean, const float* __restrict__ Wm,
    const float* __restrict__ bm, const float* __restrict__ amp,
    unsigned short* __restrict__ gm)
{
    __shared__ unsigned short Alds[128 * 64];
    __shared__ unsigned short Blds[128 * 64];
    const int r0 = blockIdx.x * 128;
    const int n0 = blockIdx.y * 128;
    const int tid = threadIdx.x;
    const float am = *amp;
    const int wave = tid >> 5, lane = tid & 31;
    const int wm = wave >> 1, wn = wave & 1;
    v8f zero = {0,0,0,0,0,0,0,0};
    v8f acc[2][4];
    for (int im = 0; im < 2; ++im)
        for (int jn = 0; jn < 4; ++jn) acc[im][jn] = zero;

    for (int kc = 0; kc < Hh; kc += 64) {
        for (int i = 0; i < 8; ++i) {
            int f = tid + i * 256;             // u16x4 index, 0..2047
            int k4 = f & 15, r = f >> 4;
            int bt = r0 + r;
            u16x4 v = {0, 0, 0, 0};
            if (bt < BT) v = *(const u16x4*)(gmean + (size_t)bt * Hh + kc + k4 * 4);
            *(u16x4*)&Alds[r * 64 + k4 * 4] = v;
        }
        for (int i = 0; i < 8; ++i) {
            int f = tid + i * 256;
            int k4 = f & 15, j = f >> 4;
            const float4 v = *(const float4*)(Wm + (size_t)(n0 + j) * Hh + kc + k4 * 4);
            u16x4 p; p.x = f2bf(v.x); p.y = f2bf(v.y); p.z = f2bf(v.z); p.w = f2bf(v.w);
            *(u16x4*)&Blds[j * 64 + k4 * 4] = p;
        }
        __syncthreads();
        for (int kk = 0; kk < 64; kk += 32) {
            v16bf a0 = frag_ld(Alds, wm * 32,      kk, 64);
            v16bf a1 = frag_ld(Alds, wm * 32 + 16, kk, 64);
            for (int jn = 0; jn < 4; ++jn) {
                v16bf bf = frag_ld(Blds, wn * 64 + jn * 16, kk, 64);
                acc[0][jn] = wmma_bf16(a0, bf, acc[0][jn]);
                acc[1][jn] = wmma_bf16(a1, bf, acc[1][jn]);
            }
        }
        __syncthreads();
    }

    const int colL = lane & 15;
    const int hiL  = (lane >> 4) & 1;
    for (int im = 0; im < 2; ++im) {
        int rbase = wm * 32 + im * 16 + hiL * 8;
        for (int jn = 0; jn < 4; ++jn) {
            int n = n0 + wn * 64 + jn * 16 + colL;
            float bias = bm[n];
            for (int v = 0; v < 8; ++v) {
                int bt = r0 + rbase + v;
                if (bt < BT) {
                    float val = prelu(acc[im][jn][v] + bias, am);
                    gm[(size_t)bt * Hh + n] = f2bf(val);
                }
            }
        }
    }
}

// ---------------- K3: GEMM3 (concat(gi,gm) -> go) + LN partial stats ----------------
// grid: (375 t-tiles of 16, 1, 4 b). block 256.
__global__ __launch_bounds__(256) void tac_k3_gemm_out(
    const unsigned short* __restrict__ gi, const unsigned short* __restrict__ gm,
    const float* __restrict__ Wo, const float* __restrict__ bo, const float* __restrict__ aop,
    float* __restrict__ go, float* __restrict__ stats)
{
    __shared__ unsigned short Alds[128 * 64];
    __shared__ unsigned short Blds[128 * 64];
    __shared__ float Dlds[128 * 128];
    __shared__ float sstat[16];
    const int t0 = blockIdx.x * 16;
    const int b  = blockIdx.z;
    const int tid = threadIdx.x;
    const float ao = *aop;
    const int wave = tid >> 5, lane = tid & 31;
    const int wm = wave >> 1, wn = wave & 1;
    v8f zero = {0,0,0,0,0,0,0,0};
    v8f acc[2][4];
    for (int im = 0; im < 2; ++im)
        for (int jn = 0; jn < 4; ++jn) acc[im][jn] = zero;
    float psum[8], psq[8];
    for (int v = 0; v < 8; ++v) { psum[v] = 0.0f; psq[v] = 0.0f; }
    if (tid < 16) sstat[tid] = 0.0f;

    for (int kc = 0; kc < 2 * Hh; kc += 64) {
        // A tile is a pure bf16 byte copy -> async DMA straight into LDS
        for (int i = 0; i < 8; ++i) {
            int f = tid + i * 256;             // 8-byte chunk index
            int k4 = f & 15, r = f >> 4;
            int tl = r >> 3, g = r & 7;
            size_t bt = (size_t)b * Tt + t0 + tl;
            const unsigned short* src = (kc < Hh)
                ? gi + (bt * Gg + g) * Hh + kc + k4 * 4
                : gm + bt * Hh + (kc - Hh) + k4 * 4;
            unsigned int lds_off = (unsigned int)(size_t)&Alds[r * 64 + k4 * 4];
            async_copy_b64(lds_off, src);
        }
        // B tile needs f32->bf16 conversion, so it goes through VGPRs
        for (int i = 0; i < 8; ++i) {
            int f = tid + i * 256;
            int k4 = f & 15, j = f >> 4;
            const float4 v = *(const float4*)(Wo + (size_t)j * (2 * Hh) + kc + k4 * 4);
            u16x4 p; p.x = f2bf(v.x); p.y = f2bf(v.y); p.z = f2bf(v.z); p.w = f2bf(v.w);
            *(u16x4*)&Blds[j * 64 + k4 * 4] = p;
        }
        async_wait0();
        __syncthreads();
        for (int kk = 0; kk < 64; kk += 32) {
            v16bf a0 = frag_ld(Alds, wm * 32,      kk, 64);
            v16bf a1 = frag_ld(Alds, wm * 32 + 16, kk, 64);
            for (int jn = 0; jn < 4; ++jn) {
                v16bf bf = frag_ld(Blds, wn * 64 + jn * 16, kk, 64);
                acc[0][jn] = wmma_bf16(a0, bf, acc[0][jn]);
                acc[1][jn] = wmma_bf16(a1, bf, acc[1][jn]);
            }
        }
        __syncthreads();
    }

    // epilogue: bias + PReLU -> stage f32 D-tile; per-g partial stats (g == v)
    const int colL = lane & 15;
    const int hiL  = (lane >> 4) & 1;
    for (int im = 0; im < 2; ++im) {
        int rbase = wm * 32 + im * 16 + hiL * 8;
        for (int jn = 0; jn < 4; ++jn) {
            int n = wn * 64 + jn * 16 + colL;
            float bias = bo[n];
            for (int v = 0; v < 8; ++v) {
                float val = prelu(acc[im][jn][v] + bias, ao);
                Dlds[(rbase + v) * 128 + n] = val;
                psum[v] += val;
                psq[v]  += val * val;
            }
        }
    }
    __syncthreads();
    // coalesced copy-out: 4096 float4 stores
    for (int i = 0; i < 16; ++i) {
        int f = tid + i * 256;
        int c4 = f & 31, r = f >> 5;
        int tl = r >> 3, g = r & 7;
        size_t btg = ((size_t)(b * Tt + t0 + tl) * Gg) + g;
        *(float4*)(go + btg * Nn + c4 * 4) = *(const float4*)&Dlds[r * 128 + c4 * 4];
    }
    // wave-level reduce then LDS, then one set of global f32 atomics
    for (int v = 0; v < 8; ++v) {
        float s = psum[v], q = psq[v];
        for (int off = 16; off > 0; off >>= 1) {
            s += __shfl_down(s, off, 32);
            q += __shfl_down(q, off, 32);
        }
        if (lane == 0) {
            atomicAdd(&sstat[v * 2 + 0], s);
            atomicAdd(&sstat[v * 2 + 1], q);
        }
    }
    __syncthreads();
    if (tid < 16) {
        int g = tid >> 1, sidx = tid & 1;
        atomicAdd(&stats[(b * Gg + g) * 2 + sidx], sstat[tid]);
    }
}

// ---------------- K4: GlobLN + transpose back + residual ----------------
// grid: (94 t-tiles of 64, 4 n-tiles of 32, 32 bg). block 256.
__global__ __launch_bounds__(256) void tac_k4_ln_out(
    const float* __restrict__ x, const float* __restrict__ go,
    const float* __restrict__ stats, const float* __restrict__ gamma,
    const float* __restrict__ beta, float* __restrict__ out)
{
    __shared__ float tile[64][36];
    const int t0 = blockIdx.x * 64;
    const int n0 = blockIdx.y * 32;
    const int bg = blockIdx.z;
    const int b = bg >> 3, g = bg & 7;
    const float cnt  = (float)Nn * (float)Tt;
    const float mean = stats[bg * 2 + 0] / cnt;
    const float var  = stats[bg * 2 + 1] / cnt - mean * mean;
    const float inv  = rsqrtf(var + EPSL);
    const int tid = threadIdx.x;

    for (int i = 0; i < 2; ++i) {
        int idx = tid + i * 256;               // 0..511 float4
        int n4 = idx & 7, tl = idx >> 3;
        int t = t0 + tl;
        float4 v = {0, 0, 0, 0};
        if (t < Tt)
            v = *(const float4*)(go + (((size_t)b * Tt + t) * Gg + g) * Nn + n0 + n4 * 4);
        tile[tl][n4 * 4 + 0] = v.x;
        tile[tl][n4 * 4 + 1] = v.y;
        tile[tl][n4 * 4 + 2] = v.z;
        tile[tl][n4 * 4 + 3] = v.w;
    }
    __syncthreads();
    for (int i = 0; i < 2; ++i) {
        int idx = tid + i * 256;
        int t4 = idx & 15, nl = idx >> 4;      // nl 0..31
        int t = t0 + t4 * 4;
        if (t < Tt) {
            int n = n0 + nl;
            float ga = gamma[n], be = beta[n];
            size_t base = ((size_t)(b * Gg + g) * Nn + n) * Tt + t;
            float4 xin = *(const float4*)(x + base);
            float4 o;
            o.x = xin.x + ga * (tile[t4 * 4 + 0][nl] - mean) * inv + be;
            o.y = xin.y + ga * (tile[t4 * 4 + 1][nl] - mean) * inv + be;
            o.z = xin.z + ga * (tile[t4 * 4 + 2][nl] - mean) * inv + be;
            o.w = xin.w + ga * (tile[t4 * 4 + 3][nl] - mean) * inv + be;
            *(float4*)(out + base) = o;
        }
    }
}

extern "C" void kernel_launch(void* const* d_in, const int* in_sizes, int n_in,
                              void* d_out, int out_size, void* d_ws, size_t ws_size,
                              hipStream_t stream) {
    (void)in_sizes; (void)n_in; (void)out_size; (void)ws_size;
    const float* x     = (const float*)d_in[0];
    const float* Wi    = (const float*)d_in[1];
    const float* bi    = (const float*)d_in[2];
    const float* ai    = (const float*)d_in[3];
    const float* Wm    = (const float*)d_in[4];
    const float* bm    = (const float*)d_in[5];
    const float* am    = (const float*)d_in[6];
    const float* Wo    = (const float*)d_in[7];
    const float* bo    = (const float*)d_in[8];
    const float* ao    = (const float*)d_in[9];
    const float* gamma = (const float*)d_in[10];
    const float* beta  = (const float*)d_in[11];
    float* outp = (float*)d_out;

    // workspace layout (bytes):
    // gi    bf16 [192000,384] : 147,456,000
    // gmean bf16 [ 24000,384] :  18,432,000
    // gm    bf16 [ 24000,384] :  18,432,000
    // go    f32  [192000,128] :  98,304,000
    // stats f32  [64]         :         256
    char* ws = (char*)d_ws;
    unsigned short* gi    = (unsigned short*)(ws);
    unsigned short* gmean = (unsigned short*)(ws + 147456000ull);
    unsigned short* gm    = (unsigned short*)(ws + 165888000ull);
    float*          go    = (float*)(ws + 184320000ull);
    float*          stats = (float*)(ws + 282624000ull);

    tac_k0_zero<<<1, 64, 0, stream>>>(stats);
    tac_k1_gemm_in<<<dim3(Tt / 16, Hh / 128, Bb), 256, 0, stream>>>(x, Wi, bi, ai, gi, gmean);
    tac_k2_gemm_mean<<<dim3((BT + 127) / 128, Hh / 128), 256, 0, stream>>>(gmean, Wm, bm, am, gm);
    tac_k3_gemm_out<<<dim3(Tt / 16, 1, Bb), 256, 0, stream>>>(gi, gm, Wo, bo, ao, go, stats);
    tac_k4_ln_out<<<dim3((Tt + 63) / 64, Nn / 32, Bb * Gg), 256, 0, stream>>>(x, go, stats, gamma, beta, outp);
}
